// MoEWithLoadBalancing_76450417869449
// MI455X (gfx1250) — matrix operand order repaired and checked by reading, not compiled, not run on hardware
//
#include <hip/hip_runtime.h>

// ---------------------------------------------------------------------------
// MoE top-1 routed GEMM for MI455X (gfx1250, wave32, WMMA).
// out[b,s,o] = sum_d x[b,s,d] * W[e(b),o,d] + bias[e(b),o]
// bf16x3 split precision: f ~ hi + lo (truncation split), f32 accumulate.
// Conversion once per element at LDS staging (v_perm_b32 packing); fragments
// read from fragment-ready bf16 planes as contiguous ds_load_b128 pairs.
// Global loads use uniform-base + 32-bit lane offset addressing.
// ---------------------------------------------------------------------------

typedef __attribute__((ext_vector_type(16))) __bf16 v16bf;
typedef __attribute__((ext_vector_type(8)))  float  v8f;

#define E_NUM   8
#define OUTD    4096
#define IND     1024
#define BATCH   16
#define SEQ     2048

#define BM      128
#define BN      128
#define BK      32
#define NKT     (IND / BK)
#define LDH     40      // LDS row stride in bf16 units (80B: b128-aligned, 16 bank groups)

// dst = (lo_src >> 16) | (hi_src & 0xffff0000)  -- one v_perm_b32
__device__ __forceinline__ unsigned pack_hi16(unsigned hi_src, unsigned lo_src) {
    return __builtin_amdgcn_perm(hi_src, lo_src, 0x07060302u);
}

// Truncation hi/lo bf16 split of 4 floats; packed stores (8B hi + 8B lo).
__device__ __forceinline__ void cvt_store(float4 v, unsigned short* hp, unsigned short* lp) {
    unsigned u0 = __builtin_bit_cast(unsigned, v.x);
    unsigned u1 = __builtin_bit_cast(unsigned, v.y);
    unsigned u2 = __builtin_bit_cast(unsigned, v.z);
    unsigned u3 = __builtin_bit_cast(unsigned, v.w);
    float l0 = v.x - __builtin_bit_cast(float, u0 & 0xffff0000u);
    float l1 = v.y - __builtin_bit_cast(float, u1 & 0xffff0000u);
    float l2 = v.z - __builtin_bit_cast(float, u2 & 0xffff0000u);
    float l3 = v.w - __builtin_bit_cast(float, u3 & 0xffff0000u);
    uint2 hv, lv;
    hv.x = pack_hi16(__builtin_bit_cast(unsigned, v.y) , u0);   // [u0.hi16, u1.hi16]
    hv.x = pack_hi16(u1, u0);
    hv.y = pack_hi16(u3, u2);
    lv.x = pack_hi16(__builtin_bit_cast(unsigned, l1), __builtin_bit_cast(unsigned, l0));
    lv.y = pack_hi16(__builtin_bit_cast(unsigned, l3), __builtin_bit_cast(unsigned, l2));
    *(uint2*)hp = hv;
    *(uint2*)lp = lv;
}

// Build a 16-element bf16 fragment from two contiguous 16B chunks in LDS.
__device__ __forceinline__ v16bf ld_frag(const unsigned short* base, int off0, int off1) {
    uint4 c0 = *(const uint4*)(base + off0);
    uint4 c1 = *(const uint4*)(base + off1);
    struct P { uint4 a, b; } p;
    p.a = c0; p.b = c1;
    return __builtin_bit_cast(v16bf, p);
}

// One k-step of fragment loads + 24 WMMAs (bf16x3: ah*bh + ah*bl + al*bh).
__device__ __forceinline__ void compute_step(const unsigned short* Ah, const unsigned short* Al,
                                             const unsigned short* Bh, const unsigned short* Bl,
                                             int waveM, int waveN, int l15, int kh8, int kh16,
                                             v8f acc[2][4]) {
    v16bf ah[2], alo[2], bh[4], blo[4];
#pragma unroll
    for (int mi = 0; mi < 2; ++mi) {
        int r = (waveM * 32 + mi * 16 + l15) * LDH;
        ah[mi]  = ld_frag(Ah + r, kh8, 16 + kh8);
        alo[mi] = ld_frag(Al + r, kh8, 16 + kh8);
    }
#pragma unroll
    for (int ni = 0; ni < 4; ++ni) {
        int r = (waveN * 64 + ni * 16 + l15) * LDH + kh8;
        bh[ni]  = ld_frag(Bh + r, 0, 16);
        blo[ni] = ld_frag(Bl + r, 0, 16);
    }
#pragma unroll
    for (int mi = 0; mi < 2; ++mi) {
#pragma unroll
        for (int ni = 0; ni < 4; ++ni) {
            acc[mi][ni] = __builtin_amdgcn_wmma_f32_16x16x32_bf16(
                false, ah[mi], false, bh[ni], (short)0, acc[mi][ni], false, false);
            acc[mi][ni] = __builtin_amdgcn_wmma_f32_16x16x32_bf16(
                false, ah[mi], false, blo[ni], (short)0, acc[mi][ni], false, false);
            acc[mi][ni] = __builtin_amdgcn_wmma_f32_16x16x32_bf16(
                false, alo[mi], false, bh[ni], (short)0, acc[mi][ni], false, false);
        }
    }
}

__global__ __launch_bounds__(256)
void moe_gemm_bf16x3_kernel(const float* __restrict__ x,
                            const float* __restrict__ probs,
                            const float* __restrict__ w,
                            const float* __restrict__ bias,
                            float* __restrict__ out) {
    __shared__ __align__(16) unsigned short Ah[BM * LDH];
    __shared__ __align__(16) unsigned short Al[BM * LDH];
    __shared__ __align__(16) unsigned short Bh[BN * LDH];
    __shared__ __align__(16) unsigned short Bl[BN * LDH];

    const int bIdx = blockIdx.z;
    const int m0   = blockIdx.y * BM;
    const int n0   = blockIdx.x * BN;

    // top-1 routing (first-max-wins, matches jnp.argmax); 8 uniform loads.
    int e = 0;
    float best = probs[bIdx * E_NUM];
#pragma unroll
    for (int j = 1; j < E_NUM; ++j) {
        float pj = probs[bIdx * E_NUM + j];
        if (pj > best) { best = pj; e = j; }
    }

    const int tid   = threadIdx.x;
    const int lane  = tid & 31;
    const int wv    = tid >> 5;
    const int waveM = wv & 3;        // 4 waves along M -> 32 rows each
    const int waveN = wv >> 2;       // 2 waves along N -> 64 cols each
    const int l15   = lane & 15;
    const int kh8   = (lane >> 4) << 3;
    const int kh16  = (lane >> 4) << 4;

    // Uniform base pointers (SGPRs); advanced by BK floats per k-step.
    const char* xb = (const char*)(x + ((size_t)bIdx * SEQ + m0) * IND);
    const char* wb = (const char*)(w + ((size_t)e * OUTD + n0) * IND);

    // Per-lane 32-bit byte offsets + LDS element offsets, computed once.
    unsigned boff[4];
    int      loff[4];
#pragma unroll
    for (int i = 0; i < 4; ++i) {
        int f   = tid + i * 256;                 // 0..1023
        int row = f >> 3;                        // 0..127
        int kq  = (f & 7) << 2;                  // float offset in 32-wide slab
        boff[i] = (unsigned)(row * IND + kq) * 4u;
        loff[i] = row * LDH + kq;
    }

    v8f acc[2][4] = {};

    // Software pipeline: prime step 0 into registers.
    float4 ra[4], rb[4];
#pragma unroll
    for (int i = 0; i < 4; ++i) {
        ra[i] = *(const float4*)(xb + boff[i]);
        rb[i] = *(const float4*)(wb + boff[i]);
    }

    for (int kt = 0; kt < NKT - 1; ++kt) {
        __syncthreads();   // previous-iteration LDS reads complete
#pragma unroll
        for (int i = 0; i < 4; ++i) {
            cvt_store(ra[i], &Ah[loff[i]], &Al[loff[i]]);
            cvt_store(rb[i], &Bh[loff[i]], &Bl[loff[i]]);
        }
        __syncthreads();   // planes visible

        xb += BK * 4;      // uniform scalar advance
        wb += BK * 4;
#pragma unroll
        for (int i = 0; i < 4; ++i) {
            ra[i] = *(const float4*)(xb + boff[i]);
            rb[i] = *(const float4*)(wb + boff[i]);
        }

        compute_step(Ah, Al, Bh, Bl, waveM, waveN, l15, kh8, kh16, acc);
    }

    // Peeled final k-step (no further global loads).
    __syncthreads();
#pragma unroll
    for (int i = 0; i < 4; ++i) {
        cvt_store(ra[i], &Ah[loff[i]], &Al[loff[i]]);
        cvt_store(rb[i], &Bh[loff[i]], &Bl[loff[i]]);
    }
    __syncthreads();
    compute_step(Ah, Al, Bh, Bl, waveM, waveN, l15, kh8, kh16, acc);

    // Epilogue: C/D layout -> element r of lane: M = r + ((lane>=16)?8:0), N = lane&15.
    const int hi8 = (lane >> 4) << 3;
#pragma unroll
    for (int mi = 0; mi < 2; ++mi) {
        int srowo = m0 + waveM * 32 + mi * 16 + hi8;
#pragma unroll
        for (int ni = 0; ni < 4; ++ni) {
            int ocol = n0 + waveN * 64 + ni * 16 + l15;
            float bv = bias[(size_t)e * OUTD + ocol];
            float* op = out + ((size_t)bIdx * SEQ + srowo) * OUTD + ocol;
#pragma unroll
            for (int r = 0; r < 8; ++r)
                op[(size_t)r * OUTD] = acc[mi][ni][r] + bv;
        }
    }
}

// Writes expert_idx (int32) into the output tail after x_out.
__global__ void moe_route_out_kernel(const float* __restrict__ probs, float* __restrict__ out) {
    int b = threadIdx.x;
    if (b < BATCH) {
        int e = 0;
        float best = probs[b * E_NUM];
#pragma unroll
        for (int j = 1; j < E_NUM; ++j) {
            float pj = probs[b * E_NUM + j];
            if (pj > best) { best = pj; e = j; }
        }
        ((int*)out)[(size_t)BATCH * SEQ * OUTD + b] = e;
    }
}

extern "C" void kernel_launch(void* const* d_in, const int* in_sizes, int n_in,
                              void* d_out, int out_size, void* d_ws, size_t ws_size,
                              hipStream_t stream) {
    const float* x     = (const float*)d_in[0];   // [16, 2048, 1024] f32
    const float* probs = (const float*)d_in[1];   // [16, 8] f32
    const float* w     = (const float*)d_in[2];   // [8, 4096, 1024] f32
    const float* bias  = (const float*)d_in[3];   // [8, 4096] f32
    float* out = (float*)d_out;                   // [16,2048,4096] f32 ++ [16] i32

    dim3 grid(OUTD / BN, SEQ / BM, BATCH);        // (32, 16, 16)
    moe_gemm_bf16x3_kernel<<<grid, 256, 0, stream>>>(x, probs, w, bias, out);
    moe_route_out_kernel<<<1, 32, 0, stream>>>(probs, out);
}
